// Isotropic_78915729096795
// MI455X (gfx1250) — matrix operand order repaired and checked by reading, not compiled, not run on hardware
//
#include <hip/hip_runtime.h>
#include <math.h>

// ---------------------------------------------------------------------------
// Types for CDNA5 (gfx1250) wave32 WMMA
// ---------------------------------------------------------------------------
typedef __bf16 bf16;
typedef __attribute__((ext_vector_type(16))) __bf16 v16bf;
typedef __attribute__((ext_vector_type(8)))  __bf16 bf16x8;
typedef __attribute__((ext_vector_type(8)))  float   v8f;

#define WMMA_BF16(a, b, c) \
  __builtin_amdgcn_wmma_f32_16x16x32_bf16(false, (a), false, (b), (short)0, (c), false, false)

// Model constants (from reference): B=2, L=2048, D=1024, H=16, Dh=64, NL=8, DFF=4096
#define MB_   2
#define LL_   2048
#define DD_   1024
#define HH_   16
#define DH_   64
#define NL_   8
#define DFF_  4096
#define MROWS (MB_ * LL_)   // 4096
#define WIN_  1024

// ---------------------------------------------------------------------------
// Fragment loaders matching the ISA wave32 VGPR layouts (CDNA5 ISA)
//   A (16x32 MxK, 16-bit): lanes 0-15 -> M=lane, K = {0..7,16..23}
//                          lanes 16-31 -> M=lane-16, K = {8..15,24..31}
//   B (32x16 KxN, 16-bit): lanes 0-15 -> N=lane, K = {0..7,16..23} ; lanes 16-31 -> K = {8..15,24..31}
// ---------------------------------------------------------------------------
static __device__ __forceinline__ v16bf load_a_frag(const bf16* __restrict__ A,
                                                    int lda, int row0, int k0) {
  const int lane = threadIdx.x & 31;
  const int row  = row0 + (lane & 15);
  const int kh   = (lane >> 4) * 8;
  const bf16* p = A + (size_t)row * lda + (k0 + kh);
  bf16x8 lo = *(const bf16x8*)p;
  bf16x8 hi = *(const bf16x8*)(p + 16);
  v16bf f;
#pragma unroll
  for (int i = 0; i < 8; ++i) { f[i] = lo[i]; f[i + 8] = hi[i]; }
  return f;
}

static __device__ __forceinline__ v16bf load_b_frag(const bf16* __restrict__ Bw,
                                                    int ldk, int col0, int k0) {
  const int lane = threadIdx.x & 31;
  const int col  = col0 + (lane & 15);
  const int kh   = (lane >> 4) * 8;
  const bf16* p = Bw + (size_t)col * ldk + (k0 + kh);
  bf16x8 lo = *(const bf16x8*)p;
  bf16x8 hi = *(const bf16x8*)(p + 16);
  v16bf f;
#pragma unroll
  for (int i = 0; i < 8; ++i) { f[i] = lo[i]; f[i + 8] = hi[i]; }
  return f;
}

// LDS tile fragment loaders (tile row stride = LDSS elements = 80B, 16B aligned)
#define LDSS 40

static __device__ __forceinline__ v16bf lds_a_frag(const bf16* t, int row0) {
  const int lane = threadIdx.x & 31;
  const bf16* p = t + (row0 + (lane & 15)) * LDSS + (lane >> 4) * 8;
  bf16x8 lo = *(const bf16x8*)p;
  bf16x8 hi = *(const bf16x8*)(p + 16);
  v16bf f;
#pragma unroll
  for (int i = 0; i < 8; ++i) { f[i] = lo[i]; f[i + 8] = hi[i]; }
  return f;
}

static __device__ __forceinline__ v16bf lds_b_frag(const bf16* t, int row0) {
  const int lane = threadIdx.x & 31;
  const bf16* p = t + (row0 + (lane & 15)) * LDSS + (lane >> 4) * 8;
  bf16x8 lo = *(const bf16x8*)p;
  bf16x8 hi = *(const bf16x8*)(p + 16);
  v16bf f;
#pragma unroll
  for (int i = 0; i < 8; ++i) { f[i] = lo[i]; f[i + 8] = hi[i]; }
  return f;
}

// One 16-byte async DMA lane-transfer: LDS[lds_off] = MEM[g]
static __device__ __forceinline__ void async_copy16(unsigned lds_off, const bf16* g) {
  asm volatile("global_load_async_to_lds_b128 %0, %1, off"
               :: "v"(lds_off), "v"(g) : "memory");
}

// ---------------------------------------------------------------------------
// GEMM: C[M,N] (f32) = A[M,K] (bf16, row-major) * W[N,K]^T (bf16, row-major)
// Block = 256 threads = 8 waves (2 M x 4 N). Block tile 128x256, wave tile
// 64x64 (16 WMMA / k-step, 128 f32 accum VGPRs). K-step 32. A/B tiles are
// double-buffered in LDS, filled with global_load_async_to_lds_b128 and
// drained with s_wait_asynccnt (CDNA5 ASYNCcnt).
// ---------------------------------------------------------------------------
__global__ __launch_bounds__(256) void gemm_bf16(const bf16* __restrict__ A,
                                                 const bf16* __restrict__ Bw,
                                                 float* __restrict__ C,
                                                 int Ndim, int Kdim) {
  __shared__ __align__(16) bf16 tA[2][128 * LDSS];
  __shared__ __align__(16) bf16 tB[2][256 * LDSS];

  const int tid  = threadIdx.x;
  const int lane = tid & 31;
  const int wid  = tid >> 5;
  const int mblk = blockIdx.y * 128;
  const int nblk = blockIdx.x * 256;

  // Async tile fill: each thread owns 2 A-chunks + 4 B-chunks of 16B per step.
  const int crow = tid >> 2;            // 0..63
  const int csub = (tid & 3) * 8;       // element offset inside 32-wide k-slice
  const bf16* gA[2];
  const bf16* gB[4];
  unsigned    lA[2], lB[4];
#pragma unroll
  for (int i = 0; i < 2; ++i) {
    gA[i] = A + (size_t)(mblk + crow + 64 * i) * Kdim + csub;
    lA[i] = (unsigned)(uintptr_t)&tA[0][(crow + 64 * i) * LDSS + csub];
  }
#pragma unroll
  for (int i = 0; i < 4; ++i) {
    gB[i] = Bw + (size_t)(nblk + crow + 64 * i) * Kdim + csub;
    lB[i] = (unsigned)(uintptr_t)&tB[0][(crow + 64 * i) * LDSS + csub];
  }
  const unsigned dA = (unsigned)(128 * LDSS * sizeof(bf16));
  const unsigned dB = (unsigned)(256 * LDSS * sizeof(bf16));

  auto issue = [&](int k0, int buf) {
    const unsigned oA = buf ? dA : 0u;
    const unsigned oB = buf ? dB : 0u;
#pragma unroll
    for (int i = 0; i < 2; ++i) async_copy16(lA[i] + oA, gA[i] + k0);
#pragma unroll
    for (int i = 0; i < 4; ++i) async_copy16(lB[i] + oB, gB[i] + k0);
  };

  const int mw = (wid >> 2) * 64;       // wave row offset within block tile
  const int nw = (wid & 3) * 64;        // wave col offset within block tile

  v8f acc[4][4];
#pragma unroll
  for (int mt = 0; mt < 4; ++mt)
#pragma unroll
    for (int nt = 0; nt < 4; ++nt)
#pragma unroll
      for (int j = 0; j < 8; ++j) acc[mt][nt][j] = 0.f;

  issue(0, 0);
  const int T = Kdim >> 5;
  for (int i = 0; i < T; ++i) {
    if (i + 1 < T) {
      issue((i + 1) << 5, (i + 1) & 1);
      asm volatile("s_wait_asynccnt 6" ::: "memory");   // previous batch done
    } else {
      asm volatile("s_wait_asynccnt 0" ::: "memory");
    }
    __syncthreads();                                    // all waves' tiles done

    const bf16* la = tA[i & 1];
    const bf16* lb = tB[i & 1];
    v16bf af[4], bfr[4];
#pragma unroll
    for (int mt = 0; mt < 4; ++mt) af[mt] = lds_a_frag(la, mw + mt * 16);
#pragma unroll
    for (int nt = 0; nt < 4; ++nt) bfr[nt] = lds_b_frag(lb, nw + nt * 16);
#pragma unroll
    for (int mt = 0; mt < 4; ++mt)
#pragma unroll
      for (int nt = 0; nt < 4; ++nt)
        acc[mt][nt] = WMMA_BF16(af[mt], bfr[nt], acc[mt][nt]);

    __syncthreads();                    // tile fully consumed before overwrite
  }

  // C/D layout: vector elem j, lanes 0-15 -> M=j, N=lane; lanes 16-31 -> M=j+8
  const int moff = 8 * (lane >> 4);
  const int ncol = nblk + nw + (lane & 15);
#pragma unroll
  for (int mt = 0; mt < 4; ++mt)
#pragma unroll
    for (int j = 0; j < 8; ++j) {
      const int m = mblk + mw + mt * 16 + j + moff;
      float* crow_p = C + (size_t)m * Ndim + ncol;
#pragma unroll
      for (int nt = 0; nt < 4; ++nt) crow_p[nt * 16] = acc[mt][nt][j];
    }
}

// ---------------------------------------------------------------------------
// Sliding-window causal flash attention. One wave per (b, h, 16-query tile).
// Q/K: [B,H,L,Dh] bf16 (rows along Dh). V: transposed [B,H,Dh,L] bf16 so that
// the attn*V B-operand is a contiguous per-lane load.
// ---------------------------------------------------------------------------
__global__ __launch_bounds__(128) void attn_kernel(const bf16* __restrict__ Q,
                                                   const bf16* __restrict__ K,
                                                   const bf16* __restrict__ Vt,
                                                   const unsigned char* __restrict__ mask,
                                                   bf16* __restrict__ O) {
  __shared__ __align__(16) bf16 pbuf[4][16 * 32];
  const int tid  = threadIdx.x;
  const int lane = tid & 31;
  const int wid  = tid >> 5;
  const int w    = blockIdx.x * 4 + wid;      // 0..4095
  const int q0   = (w & 127) * 16;
  const int h    = (w >> 7) & 15;
  const int b    = w >> 11;

  const bf16* Qh = Q  + (size_t)(b * HH_ + h) * LL_ * DH_;
  const bf16* Kh = K  + (size_t)(b * HH_ + h) * LL_ * DH_;
  const bf16* Vh = Vt + (size_t)(b * HH_ + h) * DH_ * LL_;
  const unsigned char* mrow_mask = mask + b * LL_;

  const v16bf aq0 = load_a_frag(Qh, DH_, q0, 0);
  const v16bf aq1 = load_a_frag(Qh, DH_, q0, 32);

  float mrow[8], lrow[8];
  v8f cacc[4];
#pragma unroll
  for (int j = 0; j < 8; ++j) { mrow[j] = -__builtin_inff(); lrow[j] = 0.f; }
#pragma unroll
  for (int c = 0; c < 4; ++c)
#pragma unroll
    for (int j = 0; j < 8; ++j) cacc[c][j] = 0.f;

  const int   moff  = 8 * (lane >> 4);
  const float scale = 0.125f;  // Dh^-0.5, Dh = 64
  int lo = q0 - (WIN_ - 1); if (lo < 0) lo = 0; lo &= ~31;

  for (int kbv = lo; kbv < q0 + 16; kbv += 32) {
    v8f s0, s1;
#pragma unroll
    for (int j = 0; j < 8; ++j) { s0[j] = 0.f; s1[j] = 0.f; }
    s0 = WMMA_BF16(aq0, load_b_frag(Kh, DH_, kbv,      0),  s0);
    s0 = WMMA_BF16(aq1, load_b_frag(Kh, DH_, kbv,      32), s0);
    s1 = WMMA_BF16(aq0, load_b_frag(Kh, DH_, kbv + 16, 0),  s1);
    s1 = WMMA_BF16(aq1, load_b_frag(Kh, DH_, kbv + 16, 32), s1);

    // This lane's two key columns for the whole tile (j-invariant): hoist the
    // padding-mask byte loads out of the row loop.
    const int  n0  = kbv + (lane & 15);
    const int  n1  = n0 + 16;
    const bool mk0 = mrow_mask[n0] != 0;
    const bool mk1 = mrow_mask[n1] != 0;

#pragma unroll
    for (int j = 0; j < 8; ++j) {
      const int m = q0 + j + moff;
      float p0 = s0[j] * scale;
      float p1 = s1[j] * scale;
      const bool ok0 = (n0 <= m) && (n0 >= m - (WIN_ - 1)) && mk0;
      const bool ok1 = (n1 <= m) && (n1 >= m - (WIN_ - 1)) && mk1;
      p0 = ok0 ? p0 : -__builtin_inff();
      p1 = ok1 ? p1 : -__builtin_inff();

      float rmax = fmaxf(p0, p1);            // reduce over the 16-lane row group
#pragma unroll
      for (int o = 1; o < 16; o <<= 1) rmax = fmaxf(rmax, __shfl_xor(rmax, o, 32));
      const float nm   = fmaxf(mrow[j], rmax);
      const float nms  = (nm == -__builtin_inff()) ? 0.f : nm;
      const float corr = __expf(mrow[j] - nms);
      mrow[j] = nm;
      const float e0 = __expf(p0 - nms);
      const float e1 = __expf(p1 - nms);
      float rs = e0 + e1;
#pragma unroll
      for (int o = 1; o < 16; o <<= 1) rs += __shfl_xor(rs, o, 32);
      lrow[j] = lrow[j] * corr + rs;
#pragma unroll
      for (int c = 0; c < 4; ++c) cacc[c][j] *= corr;

      const int r = j + moff;                // stage P (C-layout -> LDS row-major)
      pbuf[wid][r * 32 +      (lane & 15)] = (bf16)e0;
      pbuf[wid][r * 32 + 16 + (lane & 15)] = (bf16)e1;
    }

    asm volatile("s_wait_dscnt 0" ::: "memory");   // CDNA5 split DS counter wait

    {
      const int kh = (lane >> 4) * 8;        // re-read P as an A fragment
      const bf16* pp = &pbuf[wid][(lane & 15) * 32 + kh];
      bf16x8 lo8 = *(const bf16x8*)pp;
      bf16x8 hi8 = *(const bf16x8*)(pp + 16);
      v16bf pa;
#pragma unroll
      for (int i = 0; i < 8; ++i) { pa[i] = lo8[i]; pa[i + 8] = hi8[i]; }
#pragma unroll
      for (int c = 0; c < 4; ++c)
        cacc[c] = WMMA_BF16(pa, load_b_frag(Vh, LL_, c * 16, kbv), cacc[c]);
    }
  }

#pragma unroll
  for (int j = 0; j < 8; ++j) {
    const float  inv = 1.f / lrow[j];
    const size_t row = (size_t)(b * LL_ + q0 + j + moff);
#pragma unroll
    for (int c = 0; c < 4; ++c)
      O[row * DD_ + h * DH_ + c * 16 + (lane & 15)] = (bf16)(cacc[c][j] * inv);
  }
}

// ---------------------------------------------------------------------------
// Fused residual-add + RMSNorm.  One block per row of D=1024.
// ---------------------------------------------------------------------------
__global__ __launch_bounds__(256) void add_rmsnorm_kernel(const float* __restrict__ x,
                                                          const float* __restrict__ resin,
                                                          float* __restrict__ resout,
                                                          bf16* __restrict__ y,
                                                          const float* __restrict__ w,
                                                          int has_res) {
  const int row = blockIdx.x;
  const int t   = threadIdx.x;
  float4 v = ((const float4*)(x + (size_t)row * DD_))[t];
  if (has_res) {
    float4 r = ((const float4*)(resin + (size_t)row * DD_))[t];
    v.x += r.x; v.y += r.y; v.z += r.z; v.w += r.w;
  }
  float ssq = v.x * v.x + v.y * v.y + v.z * v.z + v.w * v.w;
#pragma unroll
  for (int o = 1; o < 32; o <<= 1) ssq += __shfl_xor(ssq, o, 32);
  __shared__ float red[8];
  if ((t & 31) == 0) red[t >> 5] = ssq;
  __syncthreads();
  float tot = 0.f;
#pragma unroll
  for (int i = 0; i < 8; ++i) tot += red[i];
  const float rstd = rsqrtf(tot * (1.f / (float)DD_) + 1e-5f);
  ((float4*)(resout + (size_t)row * DD_))[t] = v;
  float4 wv = ((const float4*)w)[t];
  bf16* yr = y + (size_t)row * DD_ + t * 4;
  yr[0] = (bf16)(v.x * rstd * wv.x);
  yr[1] = (bf16)(v.y * rstd * wv.y);
  yr[2] = (bf16)(v.z * rstd * wv.z);
  yr[3] = (bf16)(v.w * rstd * wv.w);
}

__global__ __launch_bounds__(256) void final_rmsnorm_kernel(const float* __restrict__ x,
                                                            const float* __restrict__ resin,
                                                            float* __restrict__ out,
                                                            const float* __restrict__ w) {
  const int row = blockIdx.x;
  const int t   = threadIdx.x;
  float4 v = ((const float4*)(x + (size_t)row * DD_))[t];
  float4 r = ((const float4*)(resin + (size_t)row * DD_))[t];
  v.x += r.x; v.y += r.y; v.z += r.z; v.w += r.w;
  float ssq = v.x * v.x + v.y * v.y + v.z * v.z + v.w * v.w;
#pragma unroll
  for (int o = 1; o < 32; o <<= 1) ssq += __shfl_xor(ssq, o, 32);
  __shared__ float red[8];
  if ((t & 31) == 0) red[t >> 5] = ssq;
  __syncthreads();
  float tot = 0.f;
#pragma unroll
  for (int i = 0; i < 8; ++i) tot += red[i];
  const float rstd = rsqrtf(tot * (1.f / (float)DD_) + 1e-5f);
  float4 wv = ((const float4*)w)[t];
  float4 o4;
  o4.x = v.x * rstd * wv.x;
  o4.y = v.y * rstd * wv.y;
  o4.z = v.z * rstd * wv.z;
  o4.w = v.w * rstd * wv.w;
  ((float4*)(out + (size_t)row * DD_))[t] = o4;
}

// ---------------------------------------------------------------------------
// NeoX rotary on q,k (fp32 math) + bf16 pack; V is written transposed per head.
// ---------------------------------------------------------------------------
__global__ __launch_bounds__(256) void rope_kernel(const float* __restrict__ qkv,
                                                   const float* __restrict__ cost,
                                                   const float* __restrict__ sint,
                                                   bf16* __restrict__ qb,
                                                   bf16* __restrict__ kb,
                                                   bf16* __restrict__ vtb) {
  const int t   = blockIdx.x * 256 + threadIdx.x;
  const int d   = t & 31;
  const int h   = (t >> 5) & 15;
  const int pos = (t >> 9) & (LL_ - 1);
  const int b   = t >> 20;
  const float* row = qkv + (size_t)(b * LL_ + pos) * (3 * DD_) + h * DH_;
  const float c = cost[pos * 32 + d];
  const float s = sint[pos * 32 + d];
  const float q1 = row[d],            q2 = row[32 + d];
  const float k1 = row[DD_ + d],      k2 = row[DD_ + 32 + d];
  const float v1 = row[2 * DD_ + d],  v2 = row[2 * DD_ + 32 + d];
  const size_t ob = ((size_t)(b * HH_ + h) * LL_ + pos) * DH_;
  qb[ob + d]      = (bf16)(q1 * c - q2 * s);
  qb[ob + 32 + d] = (bf16)(q2 * c + q1 * s);
  kb[ob + d]      = (bf16)(k1 * c - k2 * s);
  kb[ob + 32 + d] = (bf16)(k2 * c + k1 * s);
  const size_t vb = (size_t)(b * HH_ + h) * DH_ * LL_;
  vtb[vb + (size_t)d * LL_ + pos]        = (bf16)v1;
  vtb[vb + (size_t)(d + 32) * LL_ + pos] = (bf16)v2;
}

__global__ __launch_bounds__(256) void rope_tables_kernel(float* __restrict__ cost,
                                                          float* __restrict__ sint) {
  const int t   = blockIdx.x * 256 + threadIdx.x;  // t < 2048*32
  const int pos = t >> 5;
  const int j   = t & 31;
  const float inv_freq = powf(10000.f, -(float)j * (1.f / 32.f));
  const float a = (float)pos * inv_freq;
  cost[t] = cosf(a);
  sint[t] = sinf(a);
}

__global__ __launch_bounds__(256) void swiglu_kernel(const float* __restrict__ g,
                                                     bf16* __restrict__ u) {
  const int t = blockIdx.x * 256 + threadIdx.x;    // t < M * DFF
  const int r = t >> 12;
  const int c = t & (DFF_ - 1);
  const float* gr = g + (size_t)r * (2 * DFF_);
  const float g1 = gr[c];
  const float g2 = gr[DFF_ + c];
  const float si = g1 / (1.f + __expf(-g1));
  u[t] = (bf16)(si * g2);
}

__global__ __launch_bounds__(256) void convert_kernel(const float* __restrict__ src,
                                                      bf16* __restrict__ dst, int n) {
  const int t = blockIdx.x * 256 + threadIdx.x;
  if (t < n) dst[t] = (bf16)src[t];
}

// ---------------------------------------------------------------------------
// Host orchestration
// ---------------------------------------------------------------------------
extern "C" void kernel_launch(void* const* d_in, const int* in_sizes, int n_in,
                              void* d_out, int out_size, void* d_ws, size_t ws_size,
                              hipStream_t stream) {
  (void)in_sizes; (void)n_in; (void)out_size; (void)ws_size;
  const float*         x_in  = (const float*)d_in[0];
  const unsigned char* amask = (const unsigned char*)d_in[1];
  const float*         Wqkv  = (const float*)d_in[2];
  const float*         Wout  = (const float*)d_in[3];
  const float*         Wfc1  = (const float*)d_in[4];
  const float*         Wfc2  = (const float*)d_in[5];
  const float*         n1w   = (const float*)d_in[6];
  const float*         n2w   = (const float*)d_in[7];
  const float*         fnw   = (const float*)d_in[8];
  float*               out   = (float*)d_out;

  char*  base = (char*)d_ws;
  size_t off  = 0;
  auto alloc = [&](size_t bytes) -> char* {
    char* p = base + off;
    off += (bytes + 255) & ~(size_t)255;
    return p;
  };
  bf16*  wqkv_b = (bf16*)alloc((size_t)3 * DD_ * DD_ * 2);
  bf16*  wout_b = (bf16*)alloc((size_t)DD_ * DD_ * 2);
  bf16*  fc1_b  = (bf16*)alloc((size_t)2 * DFF_ * DD_ * 2);
  bf16*  fc2_b  = (bf16*)alloc((size_t)DD_ * DFF_ * 2);
  float* resid  = (float*)alloc((size_t)MROWS * DD_ * 4);
  float* curx   = (float*)alloc((size_t)MROWS * DD_ * 4);
  bf16*  hn     = (bf16*)alloc((size_t)MROWS * DD_ * 2);
  float* big    = (float*)alloc((size_t)MROWS * 2 * DFF_ * 4);  // qkv / attn-proj / fc1-out
  bf16*  qb     = (bf16*)alloc((size_t)MROWS * DD_ * 2);
  bf16*  kb     = (bf16*)alloc((size_t)MROWS * DD_ * 2);
  bf16*  vtb    = (bf16*)alloc((size_t)MROWS * DD_ * 2);
  bf16*  attn_o = (bf16*)alloc((size_t)MROWS * DD_ * 2);
  bf16*  ub     = (bf16*)alloc((size_t)MROWS * DFF_ * 2);
  float* cost   = (float*)alloc((size_t)LL_ * 32 * 4);
  float* sint   = (float*)alloc((size_t)LL_ * 32 * 4);

  rope_tables_kernel<<<(LL_ * 32) / 256, 256, 0, stream>>>(cost, sint);

  for (int l = 0; l < NL_; ++l) {
    convert_kernel<<<(3 * DD_ * DD_) / 256, 256, 0, stream>>>(Wqkv + (size_t)l * 3 * DD_ * DD_, wqkv_b, 3 * DD_ * DD_);
    convert_kernel<<<(DD_ * DD_) / 256, 256, 0, stream>>>(Wout + (size_t)l * DD_ * DD_, wout_b, DD_ * DD_);
    convert_kernel<<<(2 * DFF_ * DD_) / 256, 256, 0, stream>>>(Wfc1 + (size_t)l * 2 * DFF_ * DD_, fc1_b, 2 * DFF_ * DD_);
    convert_kernel<<<(DD_ * DFF_) / 256, 256, 0, stream>>>(Wfc2 + (size_t)l * DD_ * DFF_, fc2_b, DD_ * DFF_);

    // h = x (+ residual); residual = h; hn = rms(h) * norm1_w
    add_rmsnorm_kernel<<<MROWS, 256, 0, stream>>>(l == 0 ? x_in : curx, resid, resid, hn,
                                                  n1w + l * DD_, l == 0 ? 0 : 1);
    // qkv = hn @ Wqkv^T   (4096 x 3072 x 1024)
    gemm_bf16<<<dim3(3072 / 256, MROWS / 128), 256, 0, stream>>>(hn, wqkv_b, big, 3 * DD_, DD_);
    // rotary + head split + V transpose
    rope_kernel<<<(MB_ * LL_ * HH_ * 32) / 256, 256, 0, stream>>>(big, cost, sint, qb, kb, vtb);
    // sliding-window flash attention
    attn_kernel<<<(MB_ * HH_ * (LL_ / 16)) / 4, 128, 0, stream>>>(qb, kb, vtb, amask, attn_o);
    // a = ctx @ Wout^T
    gemm_bf16<<<dim3(DD_ / 256, MROWS / 128), 256, 0, stream>>>(attn_o, wout_b, big, DD_, DD_);
    // h2 = a + residual; residual = h2; hn2 = rms(h2) * norm2_w
    add_rmsnorm_kernel<<<MROWS, 256, 0, stream>>>(big, resid, resid, hn, n2w + l * DD_, 1);
    // g = hn2 @ fc1^T   (4096 x 8192 x 1024)
    gemm_bf16<<<dim3(2 * DFF_ / 256, MROWS / 128), 256, 0, stream>>>(hn, fc1_b, big, 2 * DFF_, DD_);
    // u = silu(g1) * g2
    swiglu_kernel<<<(MROWS * DFF_) / 256, 256, 0, stream>>>(big, ub);
    // x = u @ fc2^T     (4096 x 1024 x 4096)
    gemm_bf16<<<dim3(DD_ / 256, MROWS / 128), 256, 0, stream>>>(ub, fc2_b, curx, DD_, DFF_);
  }

  final_rmsnorm_kernel<<<MROWS, 256, 0, stream>>>(curx, resid, out, fnw);
}
